// MoveModule_19791209300472
// MI455X (gfx1250) — compile-verified, matrix operand order checked by reading, not run on hardware
//
#include <hip/hip_runtime.h>
#include <hip/hip_bf16.h>

#define NUM_NODES 500000
#define DIM_D 128
#define DIM_H 128
#define MOVE_DIM 4
#define K_ROWS 250000
#define NEG_SLOPE 0.01f

typedef __attribute__((ext_vector_type(16))) __bf16 v16bf;
typedef __attribute__((ext_vector_type(8)))  __bf16 v8bf;
typedef __attribute__((ext_vector_type(8)))  float  v8f;

// ---------------- zero the [N, MOVE_DIM] output ----------------
__global__ void zero_out_kernel(float4* __restrict__ out, int nvec4) {
  int i = blockIdx.x * blockDim.x + threadIdx.x;
  if (i < nvec4) out[i] = make_float4(0.f, 0.f, 0.f, 0.f);
}

// ---------------- pack W1 / padded-W2 into WMMA B-operand bf16 layout ----
// B fragment (32x16 bf16, K-step kb, N-tile nt), per lane:
//   n = lane&15 (column), half = lane>>4
//   element i (0..15): g = i>>3, j = i&7, k_local = g*16 + half*8 + j
// flat index: ((nt*4 + kb)*32 + lane)*16 + i
__global__ void pack_weights_kernel(const float* __restrict__ W1,
                                    const float* __restrict__ W2,
                                    __bf16* __restrict__ w1B,
                                    __bf16* __restrict__ w2B) {
  const int W1N = 8 * 4 * 32 * 16;   // 16384 elements
  const int W2N = 4 * 32 * 16;       // 2048 elements
  int f = blockIdx.x * blockDim.x + threadIdx.x;
  if (f < W1N) {
    int i    = f & 15;
    int lane = (f >> 4) & 31;
    int kb   = (f >> 9) & 3;
    int nt   = f >> 11;
    int n = lane & 15, half = lane >> 4;
    int g = i >> 3, j = i & 7;
    int kl = g * 16 + half * 8 + j;
    int krow = kb * 32 + kl;
    w1B[f] = (__bf16)W1[krow * DIM_H + nt * 16 + n];
  } else if (f < W1N + W2N) {
    int f2 = f - W1N;
    int i    = f2 & 15;
    int lane = (f2 >> 4) & 31;
    int kb   = f2 >> 9;
    int n = lane & 15, half = lane >> 4;
    int g = i >> 3, j = i & 7;
    int kl = g * 16 + half * 8 + j;
    int krow = kb * 32 + kl;
    w2B[f2] = (n < MOVE_DIM) ? (__bf16)W2[krow * MOVE_DIM + n] : (__bf16)0.0f;
  }
}

// ---------------- main: gather -> MLP (WMMA bf16) -> scatter ----------------
// One block = 16 gathered rows, 256 threads = 8 waves.
// Wave w computes the 16-wide N-tile [w*16, w*16+16) of h = leaky(x@W1+b1).
// Wave 0 then computes y = h@W2pad (+b2) and scatters the 4 valid columns.
__global__ __launch_bounds__(256)
void mlp_gather_scatter_kernel(const float* __restrict__ nodef,
                               const int*   __restrict__ tidx,
                               const float* __restrict__ b1,
                               const float* __restrict__ b2,
                               const __bf16* __restrict__ w1B,
                               const __bf16* __restrict__ w2B,
                               float* __restrict__ out) {
  __shared__ int    idx_lds[16];
  __shared__ __bf16 x_lds[16][DIM_D];   // gathered tile, bf16
  __shared__ __bf16 h_lds[16][DIM_H];   // layer-1 activations, bf16

  const int t = threadIdx.x;

  if (t < 16) idx_lds[t] = tidx[blockIdx.x * 16 + t];
  __syncthreads();

  // cooperative gather: each thread moves 8 consecutive f32 -> bf16 LDS
  {
    int row = t >> 4;             // 0..15
    int col = (t & 15) << 3;      // 0,8,...,120
    const float* src = nodef + (size_t)idx_lds[row] * DIM_D + col;
    float4 a = ((const float4*)src)[0];
    float4 b = ((const float4*)src)[1];
    __bf16* dst = &x_lds[row][col];
    dst[0] = (__bf16)a.x; dst[1] = (__bf16)a.y;
    dst[2] = (__bf16)a.z; dst[3] = (__bf16)a.w;
    dst[4] = (__bf16)b.x; dst[5] = (__bf16)b.y;
    dst[6] = (__bf16)b.z; dst[7] = (__bf16)b.w;
  }
  __syncthreads();

  const int lane = t & 31;
  const int wave = t >> 5;        // N-tile index nt = wave
  const int n    = lane & 15;
  const int half = lane >> 4;

  // ---- layer 1: [16x128] @ [128x16] via 4 x wmma_f32_16x16x32_bf16 ----
  v8f acc = {};
#pragma unroll
  for (int kb = 0; kb < 4; ++kb) {
    // A fragment from LDS: row M=n, K = kb*32 + {half*8+0..7, 16+half*8+0..7}
    const __bf16* ap = &x_lds[n][kb * 32 + half * 8];
    v8bf lo = *(const v8bf*)(ap);
    v8bf hi = *(const v8bf*)(ap + 16);
    v16bf a;
#pragma unroll
    for (int q = 0; q < 8; ++q) { a[q] = lo[q]; a[8 + q] = hi[q]; }
    v16bf b = ((const v16bf*)w1B)[(wave * 4 + kb) * 32 + lane];
    acc = __builtin_amdgcn_wmma_f32_16x16x32_bf16(
        false, a, false, b, (short)0, acc, false, false);
  }

  // bias + leaky relu, then stage as bf16 for layer 2
  const float bias1 = b1[wave * 16 + n];
#pragma unroll
  for (int r = 0; r < 8; ++r) {
    float v = acc[r] + bias1;
    acc[r] = (v > 0.0f) ? v : NEG_SLOPE * v;
  }
#pragma unroll
  for (int r = 0; r < 8; ++r) {
    // C/D layout: lane -> N=n, VGPR r -> M = half*8 + r
    h_lds[half * 8 + r][wave * 16 + n] = (__bf16)acc[r];
  }
  __syncthreads();

  // ---- layer 2 (wave 0 only): [16x128] @ [128x16(pad)] ----
  if (wave == 0) {
    v8f y = {};
#pragma unroll
    for (int kb = 0; kb < 4; ++kb) {
      const __bf16* ap = &h_lds[n][kb * 32 + half * 8];
      v8bf lo = *(const v8bf*)(ap);
      v8bf hi = *(const v8bf*)(ap + 16);
      v16bf a;
#pragma unroll
      for (int q = 0; q < 8; ++q) { a[q] = lo[q]; a[8 + q] = hi[q]; }
      v16bf b = ((const v16bf*)w2B)[kb * 32 + lane];
      y = __builtin_amdgcn_wmma_f32_16x16x32_bf16(
          false, a, false, b, (short)0, y, false, false);
    }
    if (n < MOVE_DIM) {
      const float bias2 = b2[n];
#pragma unroll
      for (int r = 0; r < 8; ++r) {
        int m = half * 8 + r;
        out[(size_t)idx_lds[m] * MOVE_DIM + n] = y[r] + bias2;
      }
    }
  }
}

extern "C" void kernel_launch(void* const* d_in, const int* in_sizes, int n_in,
                              void* d_out, int out_size, void* d_ws, size_t ws_size,
                              hipStream_t stream) {
  const float* nodef = (const float*)d_in[0];
  const int*   tidx  = (const int*)d_in[1];
  const float* W1    = (const float*)d_in[2];
  const float* b1    = (const float*)d_in[3];
  const float* W2    = (const float*)d_in[4];
  const float* b2    = (const float*)d_in[5];
  float* out = (float*)d_out;

  __bf16* w1B = (__bf16*)d_ws;          // 16384 bf16 = 32 KB
  __bf16* w2B = w1B + 16384;            //  2048 bf16 =  4 KB

  // 1) zero the output (N*MOVE_DIM floats, divisible by 4)
  int nvec4 = out_size / 4;
  zero_out_kernel<<<(nvec4 + 255) / 256, 256, 0, stream>>>((float4*)out, nvec4);

  // 2) pack weights into WMMA B layout (16384 + 2048 = 18432 elems)
  pack_weights_kernel<<<(18432 + 255) / 256, 256, 0, stream>>>(W1, W2, w1B, w2B);

  // 3) fused gather + MLP + scatter; K_ROWS = 16 * 15625 exactly
  mlp_gather_scatter_kernel<<<K_ROWS / 16, 256, 0, stream>>>(
      nodef, tidx, b1, b2, w1B, w2B, out);
}
